// NTXentLoss_16097537425958
// MI455X (gfx1250) — matrix operand order repaired and checked by reading, not compiled, not run on hardware
//
#include <hip/hip_runtime.h>
#include <hip/hip_bf16.h>

// ---------------------------------------------------------------------------
// NT-Xent (SimCLR) fused loss for MI455X / gfx1250, wave32 + WMMA.
//
// Roofline: GEMM = 2*8192^2*128 = 17.2 GFLOP on 4 MB of input. The 8192x8192
// sim matrix (256 MB) is never materialized; GEMM is fused with an online
// row-wise sum(exp). Logits are bounded in [-2,2] (unit vectors / TEMP=0.5),
// so exp cannot overflow -> no running-max needed for logsumexp.
// f16 A/B with f32 accumulation via v_wmma_f32_16x16x32_f16.
//
// v3 structure:
//  - Each wave owns M=32 (two 16-row tiles): one 32x16 B fragment set feeds
//    8 WMMAs instead of 4 -> halves ds_load + barrier cost per FLOP.
//  - Double-buffered 4 KB LDS column panels: next tile's global load is
//    issued before computing the current tile (latency hidden), one barrier
//    per iteration.
//  - Diag/positive handling hoisted to wave-uniform scalar branches; the
//    common path is mul+exp2+add per element (exp(2c)=exp2(c*2*log2 e)).
// ---------------------------------------------------------------------------

typedef _Float16 v16h __attribute__((ext_vector_type(16)));
typedef _Float16 v8h  __attribute__((ext_vector_type(8)));
typedef _Float16 v4h  __attribute__((ext_vector_type(4)));
typedef float    v8f  __attribute__((ext_vector_type(8)));
typedef float    v4f  __attribute__((ext_vector_type(4)));

#define B_SZ      4096
#define N_SZ      8192            // 2B rows
#define D_SZ      128
#define EPS_NRM   1e-8f
#define COL_CHUNKS 8
#define COL_TILES_TOTAL (N_SZ / 16)                    // 512
#define TILES_PER_CHUNK (COL_TILES_TOTAL / COL_CHUNKS) // 64 (power of 2)
#define WAVES_PER_BLOCK 8
#define EXP2_SCALE 2.8853900817779268f  // 2*log2(e): exp(2c)==exp2(c*this)
#define INV_TEMP   2.0f

// ---------------------------------------------------------------------------
// Kernel 1: row L2-normalize fp32 -> f16 panel in workspace.
// One wave32 per row; also zeroes the poisoned accumulators/output.
// ---------------------------------------------------------------------------
__global__ __launch_bounds__(256) void ntxent_normalize(
    const float* __restrict__ zi, const float* __restrict__ zj,
    _Float16* __restrict__ pnh, float* __restrict__ esum_ws,
    float* __restrict__ pos_ws, float* __restrict__ out)
{
  int t = blockIdx.x * blockDim.x + threadIdx.x;
  if (t < N_SZ) { esum_ws[t] = 0.0f; pos_ws[t] = 0.0f; }
  if (t == 0)   { out[0] = 0.0f; }

  int row  = t >> 5;      // one wave per row
  int lane = t & 31;
  if (row >= N_SZ) return;

  const float* src = (row < B_SZ) ? (zi + (size_t)row * D_SZ)
                                  : (zj + (size_t)(row - B_SZ) * D_SZ);
  v4f x = *(const v4f*)(src + lane * 4);
  float ss = x[0]*x[0] + x[1]*x[1] + x[2]*x[2] + x[3]*x[3];
#pragma unroll
  for (int m = 16; m >= 1; m >>= 1) ss += __shfl_xor(ss, m, 32);

  float inv = 1.0f / fmaxf(sqrtf(ss), EPS_NRM);   // torch CosineSimilarity eps
  v4h h;
  h[0] = (_Float16)(x[0] * inv);
  h[1] = (_Float16)(x[1] * inv);
  h[2] = (_Float16)(x[2] * inv);
  h[3] = (_Float16)(x[3] * inv);
  *(v4h*)(pnh + (size_t)row * D_SZ + lane * 4) = h;
}

// ---------------------------------------------------------------------------
// Kernel 2: fused tiled pn @ pn^T with online sum(exp) + positive capture.
// Grid: (32 row-blocks, 8 col-chunks), 8 waves/block; wave owns 32 rows.
//
// VGPR layouts per CDNA5 ISA 7.12.2 (16-bit, wave32):
//   A 16x32 : lane m=lane&15, hs=lane>>4; elems 0-7  -> K = 8*hs + 0..7
//                                         elems 8-15 -> K = 16 + 8*hs + 0..7
//   B 32x16 : lane n=lane&15; same K order as A (8*hs + 0..7, 16 + 8*hs + 0..7)
//   C/D 16x16 f32: lane n=lane&15; VGPR r -> row M = 8*hs + r, col N = n
// ---------------------------------------------------------------------------
__global__ __launch_bounds__(256) void ntxent_simlse(
    const _Float16* __restrict__ pnh,
    float* __restrict__ esum_ws, float* __restrict__ pos_ws)
{
  __shared__ _Float16 tileB[2][16 * D_SZ];   // double-buffered 4 KB panels

  const int lane = threadIdx.x & 31;
  const int wave = threadIdx.x >> 5;
  const int m    = lane & 15;
  const int hs   = lane >> 4;

  // wave owns row tiles rt0 = 2*(bx*8+w), rt1 = rt0+1  (32 rows)
  const int rt0      = (blockIdx.x * WAVES_PER_BLOCK + wave) * 2;
  const int rowBase0 = rt0 * 16;
  const int rowBase1 = rowBase0 + 16;
  const int growB0   = rowBase0 + 8 * hs;
  const int growB1   = rowBase1 + 8 * hs;
  const int colTile0 = blockIdx.y * TILES_PER_CHUNK;

  // wave-uniform special tiles (B_SZ multiple of 16 -> whole tile same side)
  const int diag0 = rt0,     part0 = (rowBase0 < B_SZ) ? rt0 + (B_SZ/16) : rt0 - (B_SZ/16);
  const int diag1 = rt0 + 1, part1 = (rowBase1 < B_SZ) ? rt0 + 1 + (B_SZ/16) : rt0 + 1 - (B_SZ/16);

  // --- resident A fragments: 2 tiles x 4 K-chunks (64 VGPRs) ---
  v16h A0[4], A1[4];
  {
    const _Float16* ar0 = pnh + (size_t)(rowBase0 + m) * D_SZ;
    const _Float16* ar1 = pnh + (size_t)(rowBase1 + m) * D_SZ;
#pragma unroll
    for (int kc = 0; kc < 4; ++kc) {
      const v8h lo0 = *(const v8h*)(ar0 + kc * 32 + 8 * hs);
      const v8h hi0 = *(const v8h*)(ar0 + kc * 32 + 16 + 8 * hs);
      const v8h lo1 = *(const v8h*)(ar1 + kc * 32 + 8 * hs);
      const v8h hi1 = *(const v8h*)(ar1 + kc * 32 + 16 + 8 * hs);
#pragma unroll
      for (int e = 0; e < 8; ++e) {
        A0[kc][e] = lo0[e]; A0[kc][8 + e] = hi0[e];
        A1[kc][e] = lo1[e]; A1[kc][8 + e] = hi1[e];
      }
    }
  }

  float esum0[8], esum1[8], pos0[8], pos1[8];
#pragma unroll
  for (int r = 0; r < 8; ++r) { esum0[r] = esum1[r] = pos0[r] = pos1[r] = 0.0f; }

  // staging coordinates for this thread (256 threads x one b128 = 4 KB panel)
  const int srow = threadIdx.x >> 4;          // 0..15 panel row
  const int scol = (threadIdx.x & 15) << 3;   // 0..120 f16 offset

  // prologue: stage tile 0 into buffer 0
  {
    v8h g = *(const v8h*)(pnh + (size_t)(colTile0 * 16 + srow) * D_SZ + scol);
    *(v8h*)(&tileB[0][srow * D_SZ + scol]) = g;
  }
  __syncthreads();

  for (int jt = 0; jt < TILES_PER_CHUNK; ++jt) {
    const int ct      = colTile0 + jt;
    const int colBase = ct * 16;
    const int cur     = jt & 1;

    // issue next panel's global load now; consumed after compute (latency hidden)
    const int njt = (jt + 1) & (TILES_PER_CHUNK - 1);   // wraps harmlessly at end
    v8h g = *(const v8h*)(pnh + (size_t)((colTile0 + njt) * 16 + srow) * D_SZ + scol);

    // --- 8 WMMAs: B fragment shared by both row tiles ---
    v8f c0 = {}, c1 = {};
#pragma unroll
    for (int kc = 0; kc < 4; ++kc) {
      const _Float16* bsrc = &tileB[cur][m * D_SZ + kc * 32 + 8 * hs];
      const v8h lo = *(const v8h*)(bsrc);
      const v8h hi = *(const v8h*)(bsrc + 16);
      v16h Bf;
#pragma unroll
      for (int e = 0; e < 8; ++e) { Bf[e] = lo[e]; Bf[8 + e] = hi[e]; }
      c0 = __builtin_amdgcn_wmma_f32_16x16x32_f16(false, A0[kc], false, Bf,
                                                  (short)0, c0, false, false);
      c1 = __builtin_amdgcn_wmma_f32_16x16x32_f16(false, A1[kc], false, Bf,
                                                  (short)0, c1, false, false);
    }

    // --- fused epilogue, tile 0 rows ---
    if (ct != diag0 && ct != part0) {
#pragma unroll
      for (int r = 0; r < 8; ++r)
        esum0[r] += __builtin_amdgcn_exp2f(c0[r] * EXP2_SCALE);
    } else {
      const int gcol = colBase + m;
#pragma unroll
      for (int r = 0; r < 8; ++r) {
        int grow = growB0 + r;
        int partner = (grow < B_SZ) ? (grow + B_SZ) : (grow - B_SZ);
        float e = __builtin_amdgcn_exp2f(c0[r] * EXP2_SCALE);
        if (gcol != grow)    esum0[r] += e;
        if (gcol == partner) pos0[r]   = c0[r] * INV_TEMP;
      }
    }
    // --- fused epilogue, tile 1 rows ---
    if (ct != diag1 && ct != part1) {
#pragma unroll
      for (int r = 0; r < 8; ++r)
        esum1[r] += __builtin_amdgcn_exp2f(c1[r] * EXP2_SCALE);
    } else {
      const int gcol = colBase + m;
#pragma unroll
      for (int r = 0; r < 8; ++r) {
        int grow = growB1 + r;
        int partner = (grow < B_SZ) ? (grow + B_SZ) : (grow - B_SZ);
        float e = __builtin_amdgcn_exp2f(c1[r] * EXP2_SCALE);
        if (gcol != grow)    esum1[r] += e;
        if (gcol == partner) pos1[r]   = c1[r] * INV_TEMP;
      }
    }

    // store prefetched panel into the other buffer; barrier covers both the
    // "next reads buf[1-cur]" and "this iteration done reading buf[cur]" deps
    *(v8h*)(&tileB[1 - cur][srow * D_SZ + scol]) = g;
    __syncthreads();
  }

  // reduce across the 16 lanes sharing each row (masks 1..8 stay in-group)
#pragma unroll
  for (int r = 0; r < 8; ++r) {
#pragma unroll
    for (int msk = 8; msk >= 1; msk >>= 1) {
      esum0[r] += __shfl_xor(esum0[r], msk, 32);
      pos0[r]  += __shfl_xor(pos0[r],  msk, 32);
      esum1[r] += __shfl_xor(esum1[r], msk, 32);
      pos1[r]  += __shfl_xor(pos1[r],  msk, 32);
    }
  }

  if (m == 0) {   // lanes 0 and 16 own 8 rows of each tile
#pragma unroll
    for (int r = 0; r < 8; ++r) {
      unsafeAtomicAdd(&esum_ws[growB0 + r], esum0[r]);  // global_atomic_add_f32
      unsafeAtomicAdd(&pos_ws [growB0 + r], pos0[r]);   // only 1 chunk nonzero
      unsafeAtomicAdd(&esum_ws[growB1 + r], esum1[r]);
      unsafeAtomicAdd(&pos_ws [growB1 + r], pos1[r]);
    }
  }
}

// ---------------------------------------------------------------------------
// Kernel 3: loss = sum_rows(log(sum_exp) - pos) / N
// ---------------------------------------------------------------------------
__global__ __launch_bounds__(256) void ntxent_finalize(
    const float* __restrict__ esum_ws, const float* __restrict__ pos_ws,
    float* __restrict__ out)
{
  int i = blockIdx.x * blockDim.x + threadIdx.x;
  float v = 0.0f;
  if (i < N_SZ) v = __logf(esum_ws[i]) - pos_ws[i];
#pragma unroll
  for (int m = 16; m >= 1; m >>= 1) v += __shfl_xor(v, m, 32);
  if ((threadIdx.x & 31) == 0)
    unsafeAtomicAdd(out, v * (1.0f / (float)N_SZ));
}

// ---------------------------------------------------------------------------
extern "C" void kernel_launch(void* const* d_in, const int* in_sizes, int n_in,
                              void* d_out, int out_size, void* d_ws, size_t ws_size,
                              hipStream_t stream) {
  const float* zi = (const float*)d_in[0];
  const float* zj = (const float*)d_in[1];
  float* out = (float*)d_out;

  // workspace: [ pnh f16 8192x128 (2 MB) | esum f32 8192 | pos f32 8192 ]
  _Float16* pnh   = (_Float16*)d_ws;
  float* esum_ws  = (float*)((char*)d_ws + (size_t)N_SZ * D_SZ * sizeof(_Float16));
  float* pos_ws   = esum_ws + N_SZ;

  // K1: normalize + zero accumulators (8192 waves -> 1024 blocks of 256)
  ntxent_normalize<<<(N_SZ * 32) / 256, 256, 0, stream>>>(
      zi, zj, pnh, esum_ws, pos_ws, out);

  // K2: fused WMMA GEMM + online sum(exp): 32 row-blocks x 8 col-chunks
  dim3 g2(N_SZ / 32 / WAVES_PER_BLOCK, COL_CHUNKS);  // (32, 8)
  ntxent_simlse<<<g2, WAVES_PER_BLOCK * 32, 0, stream>>>(pnh, esum_ws, pos_ws);

  // K3: finalize scalar loss
  ntxent_finalize<<<N_SZ / 256, 256, 0, stream>>>(esum_ws, pos_ws, out);
}
